// MultiHeadAttention_61813169324792
// MI455X (gfx1250) — matrix operand order repaired and checked by reading, not compiled, not run on hardware
//
#include <hip/hip_runtime.h>
#include <hip/hip_bf16.h>

// ---------------------------------------------------------------------------
// MHA forward for MI455X (gfx1250): bf16 WMMA everywhere, flash-attention,
// register-stable ping-pong pipelines inside rolled loops (no rotation movs,
// no full-unroll wait-0 collapse).
// N=2, S=T=4096, E=768, H=12, DH=64.
// ---------------------------------------------------------------------------

#define NB  2
#define SEQ 4096
#define EMB 768
#define NH  12
#define DH  64

typedef __attribute__((ext_vector_type(16))) __bf16 v16bf;
typedef __attribute__((ext_vector_type(8)))  __bf16 v8bf;
typedef __attribute__((ext_vector_type(4)))  __bf16 v4bf;
typedef __attribute__((ext_vector_type(8)))  float  v8f;
typedef __attribute__((ext_vector_type(4)))  float  v4f;

// D = A*B + C, bf16 inputs, f32 accumulate
__device__ __forceinline__ v8f wmma_bf16(v16bf a, v16bf b, v8f c) {
  return __builtin_amdgcn_wmma_f32_16x16x32_bf16(false, a, false, b,
                                                 (short)0, c, false, false);
}

// A-fragment (16x32 bf16): lane holds row (lane&15),
// one contiguous 16-elem run at k = k0+2*kbase, kbase=8*(lane>>4).
__device__ __forceinline__ v16bf load_a_bf16(const __bf16* __restrict__ row,
                                             int kbase, int k0) {
  return *(const v16bf*)(row + k0 + 2 * kbase);
}

// B-fragment: column n of B == row n of W (since B = W^T), contiguous 32 B.
__device__ __forceinline__ v16bf load_w(const __bf16* __restrict__ W,
                                        int wr, int wk) {
  return *(const v16bf*)(W + (size_t)wr * EMB + wk);
}

// ---------------------------------------------------------------------------
// Kernel 0: f32 -> bf16 conversion, 4 elems/thread
// ---------------------------------------------------------------------------
__global__ void mha_cvt_bf16(const float* __restrict__ src,
                             __bf16* __restrict__ dst, int n4) {
  int i = blockIdx.x * 256 + threadIdx.x;
  if (i < n4) {
    v4f x = ((const v4f*)src)[i];
    v4bf y;
#pragma unroll
    for (int j = 0; j < 4; ++j) y[j] = (__bf16)x[j];
    ((v4bf*)dst)[i] = y;
  }
}

// ---------------------------------------------------------------------------
// Kernel 1: projection  out = X @ W^T + b   (X bf16 [8192,768], W bf16 [768,768])
// mode 0: store bf16 as [n][h][s][d]   (Q and K)
// mode 1: store bf16 as [n][h][d][s]   (V transposed for P@V B-fragments)
// grid: (8192/128, 768/64)  block: 256 (8 waves, 16 rows each)
// ---------------------------------------------------------------------------
__global__ __launch_bounds__(256) void mha_qkv_proj(
    const __bf16* __restrict__ Xbf, const __bf16* __restrict__ Wbf,
    const float* __restrict__ bias, __bf16* __restrict__ out, int mode) {
  int lane = threadIdx.x & 31, wave = threadIdx.x >> 5;
  int half = lane >> 4, lidx = lane & 15;
  int r0 = blockIdx.x * 128 + wave * 16;
  int c0 = blockIdx.y * 64;
  int kbase = half * 8;
  int wk0 = 16 * half;
  const __bf16* xrow = Xbf + (size_t)(r0 + lidx) * EMB;

  v8f acc[4] = {v8f{}, v8f{}, v8f{}, v8f{}};

  // register-stable ping-pong: a0/b0 are loop-carried, stage 2's loads write
  // straight into them (no rotation movs); rolled loop keeps both buffers in
  // distinct registers so the scheduler can't collapse to wait-0-per-wmma.
  v16bf a0, a1, b0[4], b1[4];
  a0 = load_a_bf16(xrow, kbase, 0);
#pragma unroll
  for (int nt = 0; nt < 4; ++nt)
    b0[nt] = load_w(Wbf, c0 + nt * 16 + lidx, wk0);

#pragma unroll 1
  for (int k0 = 0; k0 < EMB - 64; k0 += 64) {
    a1 = load_a_bf16(xrow, kbase, k0 + 32);
#pragma unroll
    for (int nt = 0; nt < 4; ++nt)
      b1[nt] = load_w(Wbf, c0 + nt * 16 + lidx, k0 + 32 + wk0);
#pragma unroll
    for (int nt = 0; nt < 4; ++nt) acc[nt] = wmma_bf16(a0, b0[nt], acc[nt]);
    a0 = load_a_bf16(xrow, kbase, k0 + 64);
#pragma unroll
    for (int nt = 0; nt < 4; ++nt)
      b0[nt] = load_w(Wbf, c0 + nt * 16 + lidx, k0 + 64 + wk0);
#pragma unroll
    for (int nt = 0; nt < 4; ++nt) acc[nt] = wmma_bf16(a1, b1[nt], acc[nt]);
  }
  // epilogue: k0 = EMB-64 (last two 32-wide steps)
  a1 = load_a_bf16(xrow, kbase, EMB - 32);
#pragma unroll
  for (int nt = 0; nt < 4; ++nt)
    b1[nt] = load_w(Wbf, c0 + nt * 16 + lidx, EMB - 32 + wk0);
#pragma unroll
  for (int nt = 0; nt < 4; ++nt) acc[nt] = wmma_bf16(a0, b0[nt], acc[nt]);
#pragma unroll
  for (int nt = 0; nt < 4; ++nt) acc[nt] = wmma_bf16(a1, b1[nt], acc[nt]);

#pragma unroll
  for (int nt = 0; nt < 4; ++nt) {
    int col = c0 + nt * 16 + lidx;
    float bv = bias[col];
    int h = col >> 6, d = col & 63;
#pragma unroll
    for (int r = 0; r < 8; ++r) {
      int grow = r0 + r + 8 * half;       // C-layout row
      int nb = grow >> 12, s = grow & 4095;
      float v = acc[nt][r] + bv;
      size_t off = (mode == 0)
          ? ((((size_t)nb * NH + h) * SEQ + s) * DH + d)
          : ((((size_t)nb * NH + h) * DH + d) * SEQ + s);
      out[off] = (__bf16)v;
    }
  }
}

// ---------------------------------------------------------------------------
// Kernel 2: causal flash attention per head.
// grid: (SEQ/128, NH, NB)  block: 256 (8 waves x 16 query rows)
// Q,K: [n][h][t][64] bf16 ; V: [n][h][64][t] bf16 ; Y: [n][s][768] bf16
// ---------------------------------------------------------------------------
__global__ __launch_bounds__(256) void mha_flash_attn(
    const __bf16* __restrict__ Qbf, const __bf16* __restrict__ Kbf,
    const __bf16* __restrict__ Vt, __bf16* __restrict__ Ybuf) {
  __shared__ __bf16 lds_p[8][16 * 32];  // per-wave P tile (16x32), 8 KB total
  int lane = threadIdx.x & 31, wave = threadIdx.x >> 5;
  int half = lane >> 4, lidx = lane & 15;
  int h = blockIdx.y, nb = blockIdx.z;
  int q0 = blockIdx.x * 128 + wave * 16;
  size_t hbase = (((size_t)nb * NH + h) * SEQ) * DH;
  const __bf16* Qh = Qbf + hbase;
  const __bf16* Kh = Kbf + hbase;
  const __bf16* Vh = Vt + hbase;       // [d][t]
  int kbase = half * 8;
  int dsel = 16 * half;

  v16bf aq0 = load_a_bf16(Qh + (size_t)(q0 + lidx) * DH, kbase, 0);
  v16bf aq1 = load_a_bf16(Qh + (size_t)(q0 + lidx) * DH, kbase, 32);

  v8f o[4] = {v8f{}, v8f{}, v8f{}, v8f{}};
  float m_i[8], l_i[8];
#pragma unroll
  for (int r = 0; r < 8; ++r) { m_i[r] = -1e30f; l_i[r] = 0.f; }

  int jmax = (q0 + 15) >> 5;
  __bf16* pl = lds_p[wave];

  auto loadK = [&](int t0b, v16bf (&dst)[4]) {
    const __bf16* kp = Kh + (size_t)(t0b + lidx) * DH + dsel;
    dst[0] = *(const v16bf*)(kp);
    dst[1] = *(const v16bf*)(kp + 32);
    dst[2] = *(const v16bf*)(kp + 16 * DH);
    dst[3] = *(const v16bf*)(kp + 16 * DH + 32);
  };

  // one 32-key block; loads next block's K frags into `kbn` (hidden under
  // this block's wmma + softmax work).
  auto flash_step = [&](int t0, const v16bf (&kb)[4], v16bf (&kbn)[4],
                        bool more) {
    if (more) {
      loadK(t0 + 32, kbn);
      __builtin_prefetch(Kh + (size_t)(t0 + 64) * DH, 0, 0);
    }

    // ---- S = (Q @ K^T): two 16x16 tiles, K-dim = 64 (2 wmma each) ----
    v8f s0 = {}, s1 = {};
    s0 = wmma_bf16(aq0, kb[0], s0);
    s0 = wmma_bf16(aq1, kb[1], s0);
    s1 = wmma_bf16(aq0, kb[2], s1);
    s1 = wmma_bf16(aq1, kb[3], s1);

    // ---- issue V loads now; they drain under the softmax VALU chain ----
    v16bf vb[4];
    const __bf16* vp = Vh + (size_t)lidx * SEQ + t0 + dsel;
#pragma unroll
    for (int nt = 0; nt < 4; ++nt)
      vb[nt] = *(const v16bf*)(vp + (size_t)nt * 16 * SEQ);

    // ---- scale + causal mask (diagonal-straddling blocks only) ----
    bool need_mask = (t0 + 31) > q0;
    int tcol = t0 + lidx;
#pragma unroll
    for (int r = 0; r < 8; ++r) {
      int qr = q0 + r + 8 * half;
      float x0 = s0[r] * 0.125f;  // 1/sqrt(64)
      float x1 = s1[r] * 0.125f;
      if (need_mask) {
        if (tcol > qr)      x0 = -1e30f;
        if (tcol + 16 > qr) x1 = -1e30f;
      }
      s0[r] = x0; s1[r] = x1;
    }

    // ---- online softmax (row = per-VGPR, cols spread over lane half) ----
#pragma unroll
    for (int r = 0; r < 8; ++r) {
      float mx = fmaxf(s0[r], s1[r]);
      mx = fmaxf(mx, __shfl_xor(mx, 1));
      mx = fmaxf(mx, __shfl_xor(mx, 2));
      mx = fmaxf(mx, __shfl_xor(mx, 4));
      mx = fmaxf(mx, __shfl_xor(mx, 8));
      float mnew = fmaxf(m_i[r], mx);
      float alpha = __expf(m_i[r] - mnew);
      float p0 = __expf(s0[r] - mnew);
      float p1 = __expf(s1[r] - mnew);
      float rs = p0 + p1;
      rs += __shfl_xor(rs, 1);
      rs += __shfl_xor(rs, 2);
      rs += __shfl_xor(rs, 4);
      rs += __shfl_xor(rs, 8);
      l_i[r] = l_i[r] * alpha + rs;
      m_i[r] = mnew;
      s0[r] = p0; s1[r] = p1;
#pragma unroll
      for (int nt = 0; nt < 4; ++nt) o[nt][r] *= alpha;
    }

    // ---- P: C-layout -> A-layout via per-wave LDS tile ----
#pragma unroll
    for (int r = 0; r < 8; ++r) {
      int m = r + 8 * half;
      pl[m * 32 + lidx]      = (__bf16)s0[r];
      pl[m * 32 + 16 + lidx] = (__bf16)s1[r];
    }
    // same-wave DS ops are in-order; fence compiler + wait DS returns
    asm volatile("s_wait_dscnt 0x0" ::: "memory");
    v16bf ap = load_a_bf16(pl + (size_t)lidx * 32, kbase, 0);

    // ---- O += P(16x32) @ V(32x64): 4 wmma, vb already resident ----
#pragma unroll
    for (int nt = 0; nt < 4; ++nt) o[nt] = wmma_bf16(ap, vb[nt], o[nt]);
  };

  // ping-pong over 32-key blocks: kbA/kbB alternate roles, no rotation movs.
  v16bf kbA[4], kbB[4];
  loadK(0, kbA);
#pragma unroll 1
  for (int j = 0; j <= jmax; j += 2) {
    flash_step(j * 32, kbA, kbB, j < jmax);
    if (j + 1 <= jmax)
      flash_step(j * 32 + 32, kbB, kbA, j + 1 < jmax);
  }

  // ---- normalize and store Y as [n][s][768] bf16 ----
  float inv[8];
#pragma unroll
  for (int r = 0; r < 8; ++r) inv[r] = 1.0f / l_i[r];
#pragma unroll
  for (int nt = 0; nt < 4; ++nt) {
    int e = h * DH + nt * 16 + lidx;
#pragma unroll
    for (int r = 0; r < 8; ++r) {
      int m = r + 8 * half;
      size_t off = ((size_t)nb * SEQ + q0 + m) * EMB + e;
      Ybuf[off] = (__bf16)(o[nt][r] * inv[r]);
    }
  }
}

// ---------------------------------------------------------------------------
// Kernel 3: output projection  out = Y @ Wo^T + bo  (f32 output), pipelined
// grid: (8192/128, 768/64)  block 256
// ---------------------------------------------------------------------------
__global__ __launch_bounds__(256) void mha_out_proj(
    const __bf16* __restrict__ Ybuf, const __bf16* __restrict__ Wo,
    const float* __restrict__ bo, float* __restrict__ out) {
  int lane = threadIdx.x & 31, wave = threadIdx.x >> 5;
  int half = lane >> 4, lidx = lane & 15;
  int r0 = blockIdx.x * 128 + wave * 16;
  int c0 = blockIdx.y * 64;
  int kbase = half * 8;
  int wk0 = 16 * half;
  const __bf16* yrow = Ybuf + (size_t)(r0 + lidx) * EMB;

  v8f acc[4] = {v8f{}, v8f{}, v8f{}, v8f{}};
  v16bf a0, a1, b0[4], b1[4];
  a0 = load_a_bf16(yrow, kbase, 0);
#pragma unroll
  for (int nt = 0; nt < 4; ++nt)
    b0[nt] = load_w(Wo, c0 + nt * 16 + lidx, wk0);

#pragma unroll 1
  for (int k0 = 0; k0 < EMB - 64; k0 += 64) {
    a1 = load_a_bf16(yrow, kbase, k0 + 32);
#pragma unroll
    for (int nt = 0; nt < 4; ++nt)
      b1[nt] = load_w(Wo, c0 + nt * 16 + lidx, k0 + 32 + wk0);
#pragma unroll
    for (int nt = 0; nt < 4; ++nt) acc[nt] = wmma_bf16(a0, b0[nt], acc[nt]);
    a0 = load_a_bf16(yrow, kbase, k0 + 64);
#pragma unroll
    for (int nt = 0; nt < 4; ++nt)
      b0[nt] = load_w(Wo, c0 + nt * 16 + lidx, k0 + 64 + wk0);
#pragma unroll
    for (int nt = 0; nt < 4; ++nt) acc[nt] = wmma_bf16(a1, b1[nt], acc[nt]);
  }
  a1 = load_a_bf16(yrow, kbase, EMB - 32);
#pragma unroll
  for (int nt = 0; nt < 4; ++nt)
    b1[nt] = load_w(Wo, c0 + nt * 16 + lidx, EMB - 32 + wk0);
#pragma unroll
  for (int nt = 0; nt < 4; ++nt) acc[nt] = wmma_bf16(a0, b0[nt], acc[nt]);
#pragma unroll
  for (int nt = 0; nt < 4; ++nt) acc[nt] = wmma_bf16(a1, b1[nt], acc[nt]);

#pragma unroll
  for (int nt = 0; nt < 4; ++nt) {
    int col = c0 + nt * 16 + lidx;
    float bv = bo[col];
#pragma unroll
    for (int r = 0; r < 8; ++r) {
      int grow = r0 + r + 8 * half;
      out[(size_t)grow * EMB + col] = acc[nt][r] + bv;
    }
  }
}

// ---------------------------------------------------------------------------
// Launch
// ---------------------------------------------------------------------------
extern "C" void kernel_launch(void* const* d_in, const int* in_sizes, int n_in,
                              void* d_out, int out_size, void* d_ws, size_t ws_size,
                              hipStream_t stream) {
  const float* query = (const float*)d_in[0];
  const float* key   = (const float*)d_in[1];
  const float* value = (const float*)d_in[2];
  // d_in[3] = attn_mask (causal tril) — implemented analytically in-kernel
  const float* Wq = (const float*)d_in[4];
  const float* bq = (const float*)d_in[5];
  const float* Wk = (const float*)d_in[6];
  const float* bk = (const float*)d_in[7];
  const float* Wv = (const float*)d_in[8];
  const float* bv = (const float*)d_in[9];
  const float* Wo = (const float*)d_in[10];
  const float* bo = (const float*)d_in[11];
  float* out = (float*)d_out;

  const size_t WBYTES = (size_t)EMB * EMB * 2;                 // 1,179,648
  const size_t TBYTES = (size_t)NB * NH * SEQ * DH * 2;        // 12,582,912
  char* w = (char*)d_ws;
  __bf16* Wq_bf = (__bf16*)(w);
  __bf16* Wk_bf = (__bf16*)(w + WBYTES);
  __bf16* Wv_bf = (__bf16*)(w + 2 * WBYTES);
  __bf16* Wo_bf = (__bf16*)(w + 3 * WBYTES);
  __bf16* Qbf   = (__bf16*)(w + 4 * WBYTES);
  __bf16* Kbf   = (__bf16*)(w + 4 * WBYTES + TBYTES);
  __bf16* Vt    = (__bf16*)(w + 4 * WBYTES + 2 * TBYTES);
  __bf16* Ybuf  = (__bf16*)(w + 4 * WBYTES + 3 * TBYTES);
  __bf16* Xq    = (__bf16*)(w + 4 * WBYTES + 4 * TBYTES);
  __bf16* Xk    = (__bf16*)(w + 4 * WBYTES + 5 * TBYTES);
  __bf16* Xv    = (__bf16*)(w + 4 * WBYTES + 6 * TBYTES);

  const int WN4 = EMB * EMB / 4;
  const int XN4 = NB * SEQ * EMB / 4;
  dim3 cw((WN4 + 255) / 256), cx((XN4 + 255) / 256);
  mha_cvt_bf16<<<cw, 256, 0, stream>>>(Wq, Wq_bf, WN4);
  mha_cvt_bf16<<<cw, 256, 0, stream>>>(Wk, Wk_bf, WN4);
  mha_cvt_bf16<<<cw, 256, 0, stream>>>(Wv, Wv_bf, WN4);
  mha_cvt_bf16<<<cw, 256, 0, stream>>>(Wo, Wo_bf, WN4);
  mha_cvt_bf16<<<cx, 256, 0, stream>>>(query, Xq, XN4);
  mha_cvt_bf16<<<cx, 256, 0, stream>>>(key,   Xk, XN4);
  mha_cvt_bf16<<<cx, 256, 0, stream>>>(value, Xv, XN4);

  dim3 pg(NB * SEQ / 128, EMB / 64);
  mha_qkv_proj<<<pg, 256, 0, stream>>>(Xq, Wq_bf, bq, Qbf, 0);
  mha_qkv_proj<<<pg, 256, 0, stream>>>(Xk, Wk_bf, bk, Kbf, 0);
  mha_qkv_proj<<<pg, 256, 0, stream>>>(Xv, Wv_bf, bv, Vt,  1);

  dim3 fg(SEQ / 128, NH, NB);
  mha_flash_attn<<<fg, 256, 0, stream>>>(Qbf, Kbf, Vt, Ybuf);

  mha_out_proj<<<pg, 256, 0, stream>>>(Ybuf, Wo_bf, bo, out);
}
